// GCNJK_24481313587663
// MI455X (gfx1250) — matrix operand order repaired and checked by reading, not compiled, not run on hardware
//
#include <hip/hip_runtime.h>
#include <hip/hip_bf16.h>

// ---------------------------------------------------------------------------
// GCN (3x GCNConv + BN/ReLU/dropout) + JumpingKnowledge(max) + Linear + logsoftmax
// MI455X / gfx1250: bf16 WMMA GEMMs (v_wmma_f32_16x16x32_bf16) with per-wave
// register blocking over all column tiles (independent accumulators -> no WMMA
// RAW hazard NOPs), async global->LDS staging of the packed weight operand
// (GLOBAL_LOAD_ASYNC_TO_LDS_B128 / ASYNCcnt), f32 atomic scatter, fused
// elementwise passes emitting bf16 activations for the next GEMM.
// ---------------------------------------------------------------------------

#define N_NODES 100000
#define N_EDGES 600000
#define IN_C 128
#define HID 128
#define OUT_C 64
#define BN_EPS 1e-5f

typedef unsigned short ushort_t;
typedef __attribute__((ext_vector_type(16))) __bf16 v16bf;
typedef __attribute__((ext_vector_type(8)))  float  v8f;
typedef __attribute__((ext_vector_type(4)))  unsigned int u32x4;
typedef int v4i_vs __attribute__((vector_size(16)));   // matches async-LDS builtin param

union Frag16 { u32x4 u[2]; v16bf v; };

#if defined(__has_builtin)
#if __has_builtin(__builtin_amdgcn_global_load_async_to_lds_b128)
#define USE_ASYNC_LDS 1
#endif
#endif
#ifndef USE_ASYNC_LDS
#define USE_ASYNC_LDS 0
#endif

__device__ __forceinline__ ushort_t f2bf(float f) {
    unsigned int u = __float_as_uint(f);
    unsigned int r = u + 0x7FFFu + ((u >> 16) & 1u);   // round-to-nearest-even
    return (ushort_t)(r >> 16);
}

__device__ __forceinline__ unsigned int hash_u32(unsigned int x) {
    x ^= x >> 16; x *= 0x7FEB352Du;
    x ^= x >> 15; x *= 0x846CA68Bu;
    x ^= x >> 16;
    return x;
}

// ---------------- f32 -> bf16 row conversion (4 elem / thread) ----------------
__global__ void cvt_bf16_kernel(const float* __restrict__ in, ushort_t* __restrict__ out,
                                size_t n4) {
    size_t i = (size_t)blockIdx.x * blockDim.x + threadIdx.x;
    if (i >= n4) return;
    const float4 v = ((const float4*)in)[i];
    ushort_t* o = out + i * 4;
    o[0] = f2bf(v.x); o[1] = f2bf(v.y); o[2] = f2bf(v.z); o[3] = f2bf(v.w);
}

// ---------------- Pack W (K x Nc, f32 row-major) into per-lane WMMA B fragments
// Fragment layout (bf16 B, 32x16 K-tile): lane L -> col = L&15, khalf = L>>4;
// half h in [0,16) -> K = kt*32 + khalf*16 + h.
__global__ void pack_w_kernel(const float* __restrict__ W, ushort_t* __restrict__ Bp,
                              int kDim, int nCols) {
    int lane = threadIdx.x;                // 0..31
    int kt = blockIdx.x;                   // kDim/32 tiles
    int nt = blockIdx.y;                   // nCols/16 tiles
    int nTiles = nCols >> 4;
    int col = (nt << 4) + (lane & 15);
    int khalf = lane >> 4;
    ushort_t* out = Bp + ((size_t)(kt * nTiles + nt) * 32 + lane) * 16;
#pragma unroll
    for (int h = 0; h < 16; ++h) {
        int k = (kt << 5) + khalf * 16 + h;
        out[h] = f2bf(W[(size_t)k * nCols + col]);
    }
}

// ---------------- WMMA GEMM: C[nRows x NT*16] = A[nRows x 128](bf16) @ Bp ------
// Block = 4 waves; each wave owns one 16-row tile and register-blocks across all
// NT column tiles (NT independent accumulators). Packed B (16-32KB) is staged
// into LDS once per workgroup via GLOBAL_LOAD_ASYNC_TO_LDS_B128 (ASYNCcnt).
template <int NT>
__global__ void gemm_wmma_kernel(const ushort_t* __restrict__ A,
                                 const ushort_t* __restrict__ Bp,
                                 float* __restrict__ C,
                                 int mTiles) {
    constexpr int KT = 128 >> 5;                      // K = 128 -> 4 k-tiles
    constexpr int nCols = NT * 16;
    constexpr int BElems = KT * NT * 32 * 16;         // packed-B ushorts
    __shared__ __align__(16) ushort_t Bs[BElems];

    // ---- cooperative async staging of packed B into LDS (16B chunks) ----
    const int tid = threadIdx.y * 32 + threadIdx.x;   // 0..127
    for (int i = tid; i < BElems / 8; i += 128) {
#if USE_ASYNC_LDS
        __builtin_amdgcn_global_load_async_to_lds_b128(
            (__attribute__((address_space(1))) v4i_vs*)(Bp + (size_t)i * 8),
            (__attribute__((address_space(3))) v4i_vs*)(Bs + (size_t)i * 8),
            /*offset=*/0, /*cpol=*/0);
#else
        ((u32x4*)Bs)[i] = ((const u32x4*)Bp)[i];
#endif
    }
#if USE_ASYNC_LDS
#if __has_builtin(__builtin_amdgcn_s_wait_asynccnt)
    __builtin_amdgcn_s_wait_asynccnt(0);
#else
    asm volatile("s_wait_asynccnt 0x0" ::: "memory");
#endif
#endif
    __syncthreads();

    const int mt = blockIdx.x * 4 + threadIdx.y;      // row tile (wave-uniform)
    if (mt >= mTiles) return;                         // whole wave exits together

    const int lane = threadIdx.x;                     // 0..31
    const int khalf = lane >> 4;
    const int row = (mt << 4) + (lane & 15);
    const ushort_t* arow = A + (size_t)row * 128;

    v8f acc[NT];
#pragma unroll
    for (int nt = 0; nt < NT; ++nt) acc[nt] = (v8f)(0.0f);

#pragma unroll
    for (int kt = 0; kt < KT; ++kt) {
        // A (16-bit, 16x32): lanes 0-15 hold K = 0..15; lanes 16-31: 16..31
        Frag16 fa;
        const u32x4* pa = (const u32x4*)(arow + (kt << 5) + khalf * 16);
        fa.u[0] = pa[0];
        fa.u[1] = pa[1];
#pragma unroll
        for (int nt = 0; nt < NT; ++nt) {
            Frag16 fb;
            const u32x4* pb = (const u32x4*)(Bs + ((size_t)(kt * NT + nt) * 32 + lane) * 16);
            fb.u[0] = pb[0];
            fb.u[1] = pb[1];
            acc[nt] = __builtin_amdgcn_wmma_f32_16x16x32_bf16(
                /*neg_a=*/false, fa.v, /*neg_b=*/false, fb.v,
                /*c_mod=*/(short)0, acc[nt], /*reuse_a=*/false, /*reuse_b=*/false);
        }
    }

    // D layout: VGPR r -> M = khalf*8 + r, N = lane&15
    const int col0 = lane & 15;
    float* cbase = C + ((size_t)(mt << 4) + (size_t)khalf * 8) * nCols + col0;
#pragma unroll
    for (int nt = 0; nt < NT; ++nt)
#pragma unroll
        for (int r = 0; r < 8; ++r)
            cbase[(size_t)r * nCols + nt * 16] = acc[nt][r];
}

// ---------------- Edge scatter: AGG[dst] += w * H[src]  (32 threads / edge) ----
__global__ void scatter_kernel(const float* __restrict__ H,
                               const int* __restrict__ src, const int* __restrict__ dst,
                               const float* __restrict__ ew,
                               float* __restrict__ AGG, int nEdges) {
    long long idx = (long long)blockIdx.x * blockDim.x + threadIdx.x;
    if (idx >= (long long)nEdges * 32) return;
    int e = (int)(idx >> 5);
    int c = ((int)idx & 31) << 2;
    int s = src[e];
    int d = dst[e];
    float w = ew[e];
    if (e + 8 < nEdges)  // gfx1250: global_prefetch_b8 on a future gather row
        __builtin_prefetch(H + (size_t)src[e + 8] * HID + c, 0, 1);
    const float4 v = *(const float4*)(H + (size_t)s * HID + c);
    float* p = AGG + (size_t)d * HID + c;
    atomicAdd(p + 0, w * v.x);
    atomicAdd(p + 1, w * v.y);
    atomicAdd(p + 2, w * v.z);
    atomicAdd(p + 3, w * v.w);
}

// ---------------- Per-channel sum & sumsq of (AGG + bias) ----------------------
__global__ void col_stats_kernel(const float* __restrict__ AGG, const float* __restrict__ bias,
                                 float* __restrict__ stats, int nRows) {
    int ch = threadIdx.x;                       // 128 channels, channel-per-lane (coalesced)
    float b = bias[ch];
    float s = 0.f, s2 = 0.f;
    for (int r = blockIdx.x; r < nRows; r += gridDim.x) {
        float v = AGG[(size_t)r * HID + ch] + b;
        s += v; s2 += v * v;
    }
    atomicAdd(&stats[ch], s);
    atomicAdd(&stats[HID + ch], s2);
}

// ---------------- Fused bias + BN(train) + ReLU + dropout ----------------------
// Writes f32 JK snapshot (Xsave) and bf16 next-layer GEMM input (Xbf).
__global__ void bn_relu_drop_kernel(const float* __restrict__ AGG, const float* __restrict__ bias,
                                    const float* __restrict__ g, const float* __restrict__ bt,
                                    const float* __restrict__ stats,
                                    float* __restrict__ Xsave, ushort_t* __restrict__ Xbf,
                                    int nRows, unsigned int seed) {
    size_t i = (size_t)blockIdx.x * blockDim.x + threadIdx.x;
    size_t n = (size_t)nRows * HID;
    if (i >= n) return;
    int ch = (int)(i & (HID - 1));
    float invN = 1.0f / (float)nRows;
    float mu = stats[ch] * invN;
    float var = stats[HID + ch] * invN - mu * mu;
    float h = AGG[i] + bias[ch];
    float xn = g[ch] * (h - mu) * rsqrtf(var + BN_EPS) + bt[ch];
    float r = fmaxf(xn, 0.0f);
    Xsave[i] = r;
    unsigned int hs = hash_u32((unsigned int)i ^ seed);
    float dv = (hs & 1u) ? (r * 2.0f) : 0.0f;   // p = 0.5, scale 1/(1-p)
    Xbf[i] = f2bf(dv);
}

// ---------------- JK max over {x1, x2, agg3 + b3} -> bf16 GEMM input -----------
__global__ void jk_pack_kernel(const float* __restrict__ X1, const float* __restrict__ X2,
                               const float* __restrict__ AGG, const float* __restrict__ b3,
                               ushort_t* __restrict__ Xbf, size_t n) {
    size_t i = (size_t)blockIdx.x * blockDim.x + threadIdx.x;
    if (i >= n) return;
    int ch = (int)(i & (HID - 1));
    float v = fmaxf(fmaxf(X1[i], X2[i]), AGG[i] + b3[ch]);
    Xbf[i] = f2bf(v);
}

// ---------------- In-place bias + log_softmax (one wave32 per row of 64) -------
__global__ void bias_logsoftmax_kernel(float* __restrict__ out, const float* __restrict__ bf,
                                       int nRows) {
    int row = blockIdx.x * blockDim.y + threadIdx.y;
    if (row >= nRows) return;
    int lane = threadIdx.x;                      // 0..31 (wave32)
    float* p = out + (size_t)row * OUT_C;
    float a = p[lane] + bf[lane];
    float b = p[lane + 32] + bf[lane + 32];
    float m = fmaxf(a, b);
#pragma unroll
    for (int o = 16; o > 0; o >>= 1) m = fmaxf(m, __shfl_xor(m, o, 32));
    float se = __expf(a - m) + __expf(b - m);
#pragma unroll
    for (int o = 16; o > 0; o >>= 1) se += __shfl_xor(se, o, 32);
    float lse = m + __logf(se);
    p[lane] = a - lse;
    p[lane + 32] = b - lse;
}

// ---------------------------------------------------------------------------
extern "C" void kernel_launch(void* const* d_in, const int* in_sizes, int n_in,
                              void* d_out, int out_size, void* d_ws, size_t ws_size,
                              hipStream_t stream) {
    const float* x   = (const float*)d_in[0];
    const int*   ei  = (const int*)d_in[1];
    const float* ew  = (const float*)d_in[2];
    const float* W1  = (const float*)d_in[3];
    const float* b1  = (const float*)d_in[4];
    const float* W2  = (const float*)d_in[5];
    const float* b2  = (const float*)d_in[6];
    const float* W3  = (const float*)d_in[7];
    const float* b3  = (const float*)d_in[8];
    const float* g1  = (const float*)d_in[9];
    const float* bt1 = (const float*)d_in[10];
    const float* g2  = (const float*)d_in[11];
    const float* bt2 = (const float*)d_in[12];
    const float* Wf  = (const float*)d_in[13];
    const float* bfv = (const float*)d_in[14];
    float* outp = (float*)d_out;

    const int nE = in_sizes[1] / 2;
    const int* src = ei;
    const int* dst = ei + nE;

    const size_t NH = (size_t)N_NODES * HID;     // 12.8M elements
    const int mTiles = N_NODES / 16;             // 6250

    // ---- workspace layout (16B-aligned slices) ----
    float* X1    = (float*)d_ws;                 // NH f32
    float* X2    = X1 + NH;                      // NH f32
    float* H     = X2 + NH;                      // NH f32  (GEMM out)
    float* AGG   = H + NH;                       // NH f32  (scatter accumulator)
    float* stats = AGG + NH;                     // 256 f32 (sum | sumsq)
    ushort_t* Xbf = (ushort_t*)(stats + 256);    // NH bf16 (current GEMM A input)
    ushort_t* Wp  = Xbf + NH;                    // 128*128 bf16 (packed W fragments)

    const dim3 gemmBlk(32, 4);
    const int  gemmGrid = (mTiles + 3) / 4;      // 4 row tiles per workgroup
    const int cvtBlocks  = (int)((NH / 4 + 255) / 256);
    const int elemBlocks = (int)((NH + 255) / 256);
    const int scatBlocks = (int)(((long long)nE * 32 + 255) / 256);

    // =========================== Layer 1 ===========================
    cvt_bf16_kernel<<<cvtBlocks, 256, 0, stream>>>(x, Xbf, NH / 4);
    pack_w_kernel<<<dim3(IN_C / 32, HID / 16), 32, 0, stream>>>(W1, Wp, IN_C, HID);
    gemm_wmma_kernel<8><<<gemmGrid, gemmBlk, 0, stream>>>(Xbf, Wp, H, mTiles);
    (void)hipMemsetAsync(AGG, 0, NH * sizeof(float), stream);
    scatter_kernel<<<scatBlocks, 256, 0, stream>>>(H, src, dst, ew, AGG, nE);
    (void)hipMemsetAsync(stats, 0, 256 * sizeof(float), stream);
    col_stats_kernel<<<512, HID, 0, stream>>>(AGG, b1, stats, N_NODES);
    bn_relu_drop_kernel<<<elemBlocks, 256, 0, stream>>>(AGG, b1, g1, bt1, stats,
                                                        X1, Xbf, N_NODES, 0x9E3779B9u);

    // =========================== Layer 2 ===========================
    pack_w_kernel<<<dim3(HID / 32, HID / 16), 32, 0, stream>>>(W2, Wp, HID, HID);
    gemm_wmma_kernel<8><<<gemmGrid, gemmBlk, 0, stream>>>(Xbf, Wp, H, mTiles);
    (void)hipMemsetAsync(AGG, 0, NH * sizeof(float), stream);
    scatter_kernel<<<scatBlocks, 256, 0, stream>>>(H, src, dst, ew, AGG, nE);
    (void)hipMemsetAsync(stats, 0, 256 * sizeof(float), stream);
    col_stats_kernel<<<512, HID, 0, stream>>>(AGG, b2, stats, N_NODES);
    bn_relu_drop_kernel<<<elemBlocks, 256, 0, stream>>>(AGG, b2, g2, bt2, stats,
                                                        X2, Xbf, N_NODES, 0x85EBCA6Bu);

    // =========================== Layer 3 ===========================
    pack_w_kernel<<<dim3(HID / 32, HID / 16), 32, 0, stream>>>(W3, Wp, HID, HID);
    gemm_wmma_kernel<8><<<gemmGrid, gemmBlk, 0, stream>>>(Xbf, Wp, H, mTiles);
    (void)hipMemsetAsync(AGG, 0, NH * sizeof(float), stream);
    scatter_kernel<<<scatBlocks, 256, 0, stream>>>(H, src, dst, ew, AGG, nE);

    // ============== JK max -> final Linear -> log_softmax ==========
    jk_pack_kernel<<<elemBlocks, 256, 0, stream>>>(X1, X2, AGG, b3, Xbf, NH);
    pack_w_kernel<<<dim3(HID / 32, OUT_C / 16), 32, 0, stream>>>(Wf, Wp, HID, OUT_C);
    gemm_wmma_kernel<4><<<gemmGrid, gemmBlk, 0, stream>>>(Xbf, Wp, outp, mTiles);
    bias_logsoftmax_kernel<<<(N_NODES + 7) / 8, dim3(32, 8), 0, stream>>>(outp, bfv, N_NODES);
}
